// Sub2GeneDifferCrossAttn_24429773980043
// MI455X (gfx1250) — matrix operand order repaired and checked by reading, not compiled, not run on hardware
//
#include <hip/hip_runtime.h>
#include <hip/hip_bf16.h>

// ---------------------------------------------------------------------------
// Sub2GeneDifferCrossAttn for MI455X (gfx1250), wave32 + WMMA bf16 pipeline.
// Round 3: TDM staging of the Q tile using the 6-arg tensor_load_to_lds
// builtin (clang-23 / therock-10.0 signature) + s_wait_tensorcnt.
// ---------------------------------------------------------------------------

typedef __bf16 bf16_t;
typedef __attribute__((ext_vector_type(16))) __bf16 v16bf;
typedef __attribute__((ext_vector_type(8)))  __bf16 v8bf;
typedef __attribute__((ext_vector_type(8)))  float  v8f;
typedef unsigned int u32x4 __attribute__((ext_vector_type(4)));
typedef int          i32x8 __attribute__((ext_vector_type(8)));
typedef int          i32x4 __attribute__((ext_vector_type(4)));

#define B_   8
#define S_   512
#define E_   768
#define GE_  32
#define HD_  384
#define L_   4096
#define LAMBDA_INIT 0.2f

// D = A(16x32 bf16) * B(32x16 bf16) + C(16x16 f32)
__device__ __forceinline__ v8f wmma_bf16(v16bf a, v16bf b, v8f c) {
  return __builtin_amdgcn_wmma_f32_16x16x32_bf16(false, a, false, b,
                                                 (short)0, c, false, false);
}

// A fragment (16-bit A 16x32): lane m = lane&15,
// VGPR0-3 hold K = koff+0..7, VGPR4-7 hold K = koff+8..15, koff = lane<16?0:16.
__device__ __forceinline__ v16bf load_a_frag(const bf16_t* base, int stride, int lane) {
  const int m    = lane & 15;
  const int koff = (lane < 16) ? 0 : 16;
  const bf16_t* p = base + m * stride + koff;
  v8bf lo = *(const v8bf*)(p);
  v8bf hi = *(const v8bf*)(p + 8);
  v16bf a;
#pragma unroll
  for (int i = 0; i < 8; ++i) { a[i] = lo[i]; a[8 + i] = hi[i]; }
  return a;
}

// B fragment (32x16): lanes 0-15 hold K=0-15 of column N=lane,
// lanes 16-31 hold K=16-31 of column N=lane-16.  Storage is N-major
// (element (k,n) at base + n*stride + k) so the 16 K values are contiguous.
__device__ __forceinline__ v16bf load_b_frag(const bf16_t* base, int stride, int lane) {
  const int n    = lane & 15;
  const int koff = (lane < 16) ? 0 : 16;
  return *(const v16bf*)(base + n * stride + koff);
}

// ---------------------------------------------------------------------------
// Generic 64x64-tile bf16 WMMA GEMM:  C[M,N] = A[M,K] * B[K,N] * scale
// 256 threads = 8 wave32; each wave owns one 16-row strip and two 16-col tiles.
// ---------------------------------------------------------------------------
template<bool A_BF16, bool B_BF16, bool OUT_BF16>
__global__ void gemm_wmma_kernel(const void* __restrict__ Ap,
                                 const void* __restrict__ Bp,
                                 void* __restrict__ Cp,
                                 int M, int N, int Kd,
                                 long long sA, long long sB, long long sC,
                                 float scale)
{
  __shared__ alignas(32) bf16_t As[64 * 32];   // [m][k]
  __shared__ alignas(32) bf16_t Bs[64 * 32];   // transposed: [n][k]

  const int tid  = threadIdx.x;
  const int lane = tid & 31;
  const int wave = tid >> 5;
  const int n0   = blockIdx.x * 64;
  const int m0   = blockIdx.y * 64;
  const int bz   = blockIdx.z;

  const long long aBase = (long long)bz * sA;
  const long long bBase = (long long)bz * sB;
  const long long cBase = (long long)bz * sC;

  const int mi  = wave >> 1;        // 0..3 : which 16-row strip
  const int ni0 = (wave & 1) * 2;   // 0 or 2 : first of two 16-col tiles

  v8f acc0 = {};
  v8f acc1 = {};

  const int ar = tid >> 2;          // 0..63
  const int ac = (tid & 3) * 8;     // 0,8,16,24
  const int br = tid >> 3;          // 0..31
  const int bc = (tid & 7) * 8;     // 0..56

  for (int k0 = 0; k0 < Kd; k0 += 32) {
    // ---- stage A chunk [64x32] ----
    {
      const long long off = aBase + (long long)(m0 + ar) * Kd + (k0 + ac);
      if constexpr (A_BF16) {
        *(v8bf*)(&As[ar * 32 + ac]) = *(const v8bf*)((const bf16_t*)Ap + off);
      } else {
        v8f v = *(const v8f*)((const float*)Ap + off);
        v8bf w;
#pragma unroll
        for (int j = 0; j < 8; ++j) w[j] = (bf16_t)v[j];
        *(v8bf*)(&As[ar * 32 + ac]) = w;
      }
      if (k0 + 32 < Kd)   // prefetch next A chunk -> global_prefetch_b8
        __builtin_prefetch((const char*)Ap + (off + 32) * (A_BF16 ? 2 : 4), 0, 1);
    }
    // ---- stage B chunk [32x64], transposed to N-major ----
    {
      const long long off = bBase + (long long)(k0 + br) * N + (n0 + bc);
      if constexpr (B_BF16) {
        v8bf v = *(const v8bf*)((const bf16_t*)Bp + off);
#pragma unroll
        for (int j = 0; j < 8; ++j) Bs[(bc + j) * 32 + br] = v[j];
      } else {
        v8f v = *(const v8f*)((const float*)Bp + off);
#pragma unroll
        for (int j = 0; j < 8; ++j) Bs[(bc + j) * 32 + br] = (bf16_t)v[j];
      }
    }
    __syncthreads();

    v16bf a  = load_a_frag(&As[(mi * 16) * 32], 32, lane);
    v16bf b0 = load_b_frag(&Bs[(ni0 * 16) * 32], 32, lane);
    v16bf b1 = load_b_frag(&Bs[((ni0 + 1) * 16) * 32], 32, lane);
    acc0 = wmma_bf16(a, b0, acc0);
    acc1 = wmma_bf16(a, b1, acc1);
    __syncthreads();
  }

  // C/D layout: VGPR r -> (M = r + (lane<16?0:8), N = lane&15)
  const int col  = n0 + (lane & 15);
  const int row0 = m0 + mi * 16 + ((lane < 16) ? 0 : 8);
#pragma unroll
  for (int r = 0; r < 8; ++r) {
    const long long o0 = cBase + (long long)(row0 + r) * N + col + ni0 * 16;
    const long long o1 = cBase + (long long)(row0 + r) * N + col + (ni0 + 1) * 16;
    const float v0 = acc0[r] * scale;
    const float v1 = acc1[r] * scale;
    if constexpr (OUT_BF16) {
      ((bf16_t*)Cp)[o0] = (bf16_t)v0;
      ((bf16_t*)Cp)[o1] = (bf16_t)v1;
    } else {
      ((float*)Cp)[o0] = v0;
      ((float*)Cp)[o1] = v1;
    }
  }
}

// ---------------------------------------------------------------------------
// lambda = exp(lq1.lk1) - exp(lq2.lk2) + LAMBDA_INIT
// ---------------------------------------------------------------------------
__global__ void lambda_kernel(const float* __restrict__ lq1, const float* __restrict__ lk1,
                              const float* __restrict__ lq2, const float* __restrict__ lk2,
                              float* __restrict__ lamp)
{
  __shared__ float r1[256], r2[256];
  float p1 = 0.f, p2 = 0.f;
  for (int i = threadIdx.x; i < HD_; i += 256) {
    p1 += lq1[i] * lk1[i];
    p2 += lq2[i] * lk2[i];
  }
  r1[threadIdx.x] = p1; r2[threadIdx.x] = p2;
  __syncthreads();
  for (int d = 128; d > 0; d >>= 1) {
    if (threadIdx.x < d) { r1[threadIdx.x] += r1[threadIdx.x + d];
                           r2[threadIdx.x] += r2[threadIdx.x + d]; }
    __syncthreads();
  }
  if (threadIdx.x == 0) *lamp = __expf(r1[0]) - __expf(r2[0]) + LAMBDA_INIT;
}

// ---------------------------------------------------------------------------
// Fused scores + differential softmax + dw.
// One workgroup = (b, 16 S-rows).  280 KB dynamic LDS holds the Q tile and the
// full 16 x 4096 score rows for BOTH heads (CDNA5 320KB LDS), so softmax stats
// and the global row-min of dw are computed without spilling scores to HBM.
// The Q tile (24 KB, contiguous) is staged by the Tensor Data Mover:
// one 1-D D# (data_size=8B, 3072 elements), tracked by TENSORcnt.
// ---------------------------------------------------------------------------
__global__ void scores_softmax_kernel(const bf16_t* __restrict__ Qbf,
                                      const bf16_t* __restrict__ Kbf,
                                      const int* __restrict__ qmask,
                                      const int* __restrict__ kmask,
                                      const float* __restrict__ lamp,
                                      float* __restrict__ dw_out,
                                      bf16_t* __restrict__ dwbf)
{
  extern __shared__ __align__(64) char smem[];
  bf16_t* qtile = (bf16_t*)smem;          // [16][768]
  bf16_t* sc0   = qtile + 16 * E_;        // [16][4096]
  bf16_t* sc1   = sc0 + 16 * L_;          // [16][4096]

  const int tid  = threadIdx.x;
  const int lane = tid & 31;
  const int wave = tid >> 5;
  const int b    = blockIdx.x >> 5;           // S_/16 = 32 blocks per batch
  const int s0   = (blockIdx.x & 31) * 16;

  // ---- TDM-stage Q tile (pre-scaled by HD^-0.5 at projection time) ----
  const bf16_t* qsrc = Qbf + ((long long)b * S_ + s0) * E_;
  if (wave == 0) {
    // D# group 0: count=1 | lds_addr | global_addr[56:0] | type=2
    const unsigned lds_addr = (unsigned)(size_t)qtile;   // flat addr[31:0] == LDS byte offset
    const unsigned long long ga = (unsigned long long)(size_t)qsrc;
    u32x4 g0;
    g0[0] = 1u;                                          // count=1, user mode
    g0[1] = lds_addr;
    g0[2] = (unsigned)(ga & 0xFFFFFFFFu);
    g0[3] = (unsigned)((ga >> 32) & 0x01FFFFFFu) | (2u << 30);  // type=2 ("image")
    // D# group 1: data_size=8B; tensor_dim0=tile_dim0=3072 (24KB/8B), 1 row.
    i32x8 g1;
    g1[0] = (int)(3u << 16);          // workgroup_mask=0, data_size=3 (8B)
    g1[1] = (int)(0xC00u << 16);      // tensor_dim0[15:0] = 3072 (bits 79:48)
    g1[2] = (int)(1u << 16);          // tensor_dim0 hi=0, tensor_dim1 lo = 1 (bits 111:80)
    g1[3] = (int)(0xC00u << 16);      // tensor_dim1 hi=0, tile_dim0 = 3072 (bits 127:112)
    g1[4] = 0;                        // tile_dim1=0 (unused), tile_dim2=0
    g1[5] = 3072;                     // tensor_dim0_stride lo (bits 207:160)
    g1[6] = 0;
    g1[7] = 0;
    i32x4 z4 = {0, 0, 0, 0};          // groups 2/3: dims 2..4 unused
    i32x8 z8 = {0, 0, 0, 0, 0, 0, 0, 0};
    __builtin_amdgcn_tensor_load_to_lds(g0, g1, z4, z4, z8, 0);
    __builtin_amdgcn_s_wait_tensorcnt((short)0);
  }
  __syncthreads();

  // ---- scores via WMMA, masked, into LDS (bf16) ----
  for (int h = 0; h < 2; ++h) {
    const int hbase = h * HD_;
    bf16_t* sc = h ? sc1 : sc0;
    for (int t = wave; t < L_ / 16; t += 8) {
      const int l0 = t * 16;
      v8f c = {};
      // Kbf is [b][l][e]: per-lane K values contiguous in e (N-major form).
      const bf16_t* kb = Kbf + ((long long)b * L_ + l0 + (lane & 15)) * E_
                             + hbase + ((lane < 16) ? 0 : 16);
#pragma unroll
      for (int kc = 0; kc < HD_ / 32; ++kc) {   // 12 WMMA per tile
        v16bf a  = load_a_frag(qtile + hbase + kc * 32, E_, lane);
        v16bf bv = *(const v16bf*)(kb + kc * 32);
        c = wmma_bf16(a, bv, c);
      }
      const int n     = lane & 15;
      const int mrow0 = (lane < 16) ? 0 : 8;
      const int km    = kmask[b * L_ + l0 + n];
#pragma unroll
      for (int r = 0; r < 8; ++r) {
        const int m = mrow0 + r;
        float v = c[r];
        if (!qmask[b * S_ + s0 + m] || !km) v = -1e20f;
        sc[m * L_ + l0 + n] = (bf16_t)v;
      }
    }
  }
  __syncthreads();

  // ---- per-row differential softmax; wave w owns rows 2w, 2w+1 ----
  const float lam = *lamp;
  for (int rr = 0; rr < 2; ++rr) {
    const int m = wave * 2 + rr;
    const int s = s0 + m;
    const bf16_t* r0 = sc0 + m * L_;
    const bf16_t* r1 = sc1 + m * L_;

    float mx0 = -3.4e38f, mx1 = -3.4e38f;
    for (int j = lane; j < L_; j += 32) {
      mx0 = fmaxf(mx0, (float)r0[j]);
      mx1 = fmaxf(mx1, (float)r1[j]);
    }
#pragma unroll
    for (int d = 16; d > 0; d >>= 1) {
      mx0 = fmaxf(mx0, __shfl_xor(mx0, d, 32));
      mx1 = fmaxf(mx1, __shfl_xor(mx1, d, 32));
    }
    float sm0 = 0.f, sm1 = 0.f;
    for (int j = lane; j < L_; j += 32) {
      sm0 += __expf((float)r0[j] - mx0);
      sm1 += __expf((float)r1[j] - mx1);
    }
#pragma unroll
    for (int d = 16; d > 0; d >>= 1) {
      sm0 += __shfl_xor(sm0, d, 32);
      sm1 += __shfl_xor(sm1, d, 32);
    }
    const float inv0 = 1.f / (sm0 + 1e-8f);
    const float inv1 = lam / (sm1 + 1e-8f);

    float mn = 3.4e38f;
    for (int j = lane; j < L_; j += 32) {
      const float v = __expf((float)r0[j] - mx0) * inv0
                    - __expf((float)r1[j] - mx1) * inv1;
      mn = fminf(mn, v);
    }
#pragma unroll
    for (int d = 16; d > 0; d >>= 1) mn = fminf(mn, __shfl_xor(mn, d, 32));

    const int qm = qmask[b * S_ + s];
    float*  orow = dw_out + ((long long)b * S_ + s) * L_;
    bf16_t* brow = dwbf   + ((long long)b * S_ + s) * L_;
    for (int j = lane; j < L_; j += 32) {
      float v = __expf((float)r0[j] - mx0) * inv0
              - __expf((float)r1[j] - mx1) * inv1 - mn + 1e-5f;
      if (!qm || !kmask[b * L_ + j]) v = 0.f;
      orow[j] = v;
      brow[j] = (bf16_t)v;
    }
  }
}

// ---------------------------------------------------------------------------
// RMSNorm over E, fused with the (1 - lambda_init) output scale; emits bf16.
// ---------------------------------------------------------------------------
__global__ void rmsnorm_kernel(const float* __restrict__ o,
                               const float* __restrict__ rms_w,
                               bf16_t* __restrict__ onbf)
{
  __shared__ float red[256];
  const long long row = blockIdx.x;
  const float* src = o + row * E_;
  float acc = 0.f;
  for (int e = threadIdx.x; e < E_; e += 256) { const float v = src[e]; acc += v * v; }
  red[threadIdx.x] = acc;
  __syncthreads();
  for (int d = 128; d > 0; d >>= 1) {
    if (threadIdx.x < d) red[threadIdx.x] += red[threadIdx.x + d];
    __syncthreads();
  }
  const float rs = rsqrtf(red[0] / (float)E_ + 1e-5f) * (1.0f - LAMBDA_INIT);
  for (int e = threadIdx.x; e < E_; e += 256)
    onbf[row * E_ + e] = (bf16_t)(src[e] * rs * rms_w[e]);
}

// ---------------------------------------------------------------------------
extern "C" void kernel_launch(void* const* d_in, const int* in_sizes, int n_in,
                              void* d_out, int out_size, void* d_ws, size_t ws_size,
                              hipStream_t stream)
{
  const float* query = (const float*)d_in[0];   // [B,S,E]
  const float* key   = (const float*)d_in[1];   // [B,P,G,GE] -> [B*L, GE]
  const int*   qmask = (const int*)d_in[2];     // [B,S]
  const int*   kmask = (const int*)d_in[3];     // [B,P,G] -> [B,L]
  const float* Wq    = (const float*)d_in[4];   // [E,E]
  const float* Wk    = (const float*)d_in[5];   // [GE,E]
  const float* Wv    = (const float*)d_in[6];   // [GE,E]
  const float* Wout  = (const float*)d_in[7];   // [E,E]
  const float* lq1   = (const float*)d_in[8];
  const float* lk1   = (const float*)d_in[9];
  const float* lq2   = (const float*)d_in[10];
  const float* lk2   = (const float*)d_in[11];
  const float* rms_w = (const float*)d_in[12];

  // ---- workspace layout ----
  char* ws = (char*)d_ws;
  size_t off = 0;
  float*  lamp = (float*)(ws + off);  off += 256;
  bf16_t* Qbf  = (bf16_t*)(ws + off); off += (size_t)B_ * S_ * E_ * 2;  // 6.3 MB
  bf16_t* Kbf  = (bf16_t*)(ws + off); off += (size_t)B_ * L_ * E_ * 2;  // 50.3 MB
  bf16_t* Vbf  = (bf16_t*)(ws + off); off += (size_t)B_ * L_ * E_ * 2;  // 50.3 MB
  bf16_t* dwbf = (bf16_t*)(ws + off); off += (size_t)B_ * S_ * L_ * 2;  // 33.6 MB
  float*  of32 = (float*)(ws + off);  off += (size_t)B_ * S_ * E_ * 4;  // 12.6 MB
  bf16_t* onbf = (bf16_t*)(ws + off); off += (size_t)B_ * S_ * E_ * 2;  // 6.3 MB

  float* out_main = (float*)d_out;                       // [B,S,E]
  float* dw_out   = out_main + (size_t)B_ * S_ * E_;     // [B,S,L]

  // 0) lambda scalar
  lambda_kernel<<<1, 256, 0, stream>>>(lq1, lk1, lq2, lk2, lamp);

  // 1) Q = query @ Wq, scaled by HD^-0.5, -> bf16
  gemm_wmma_kernel<false, false, true>
      <<<dim3(E_ / 64, (B_ * S_) / 64, 1), 256, 0, stream>>>(
      query, Wq, Qbf, B_ * S_, E_, E_, 0, 0, 0, 0.05103103630798288f);

  // 2) Kf = key @ Wk, Vf = key @ Wv  (K-dim GE=32 -> single WMMA chunk)
  gemm_wmma_kernel<false, false, true>
      <<<dim3(E_ / 64, (B_ * L_) / 64, 1), 256, 0, stream>>>(
      key, Wk, Kbf, B_ * L_, E_, GE_, 0, 0, 0, 1.0f);
  gemm_wmma_kernel<false, false, true>
      <<<dim3(E_ / 64, (B_ * L_) / 64, 1), 256, 0, stream>>>(
      key, Wv, Vbf, B_ * L_, E_, GE_, 0, 0, 0, 1.0f);

  // 3) fused scores + differential softmax + dw (280KB dynamic LDS)
  const size_t smem_sz = (size_t)16 * E_ * 2 + 2 * (size_t)16 * L_ * 2;  // 286720 B
  scores_softmax_kernel<<<B_ * (S_ / 16), 256, smem_sz, stream>>>(
      Qbf, Kbf, qmask, kmask, lamp, dw_out, dwbf);

  // 4) o = dw @ Vf  (batched over B)
  gemm_wmma_kernel<true, true, false>
      <<<dim3(E_ / 64, S_ / 64, B_), 256, 0, stream>>>(
      dwbf, Vbf, of32, S_, E_, L_,
      (long long)S_ * L_, (long long)L_ * E_, (long long)S_ * E_, 1.0f);

  // 5) RMSNorm * rms_w * (1 - lambda_init) -> bf16
  rmsnorm_kernel<<<B_ * S_, 256, 0, stream>>>(of32, rms_w, onbf);

  // 6) out = o_n @ Wout -> f32 d_out
  gemm_wmma_kernel<true, false, false>
      <<<dim3(E_ / 64, (B_ * S_) / 64, 1), 256, 0, stream>>>(
      onbf, Wout, out_main, B_ * S_, E_, E_, 0, 0, 0, 1.0f);
}
